// CSTM_29205777612976
// MI455X (gfx1250) — matrix operand order repaired and checked by reading, not compiled, not run on hardware
//
#include <hip/hip_runtime.h>
#include <hip/hip_bf16.h>
#include <stdint.h>

// ---------------------------------------------------------------------------
// Types matching the CDNA5 WMMA builtin signatures
// ---------------------------------------------------------------------------
typedef __attribute__((ext_vector_type(16))) __bf16 v16bf;
typedef __attribute__((ext_vector_type(8)))  __bf16 v8bf;
typedef __attribute__((ext_vector_type(8)))  float  v8f;

#define C_CH   256
#define HH     56
#define WWID   56
#define HWSZ   3136
#define T_LEN  8
#define NT_TOT 64

#define M_TILE 128        // co per workgroup
#define N_TILE 128        // pixels per workgroup (last block padded: 25*128=3200)
#define K_BLK  32
#define LDS_STRIDE 40     // bf16 units per row (80B: 16B aligned, bank-spread)
#define N_BLOCKS 25       // ceil(3136/128)

// d_ws layout in uint32 units
#define W1P_OFF 0                         // 3*256*256
#define W2P_OFF (3 * 256 * 256)           // 9*256*256
#define YP_OFF  (W2P_OFF + 9 * 256 * 256) // 64*256*3136

// ---------------------------------------------------------------------------
// bf16 helpers (bit-level, RNE)
// ---------------------------------------------------------------------------
static __device__ __forceinline__ uint16_t f2bf_bits(float f) {
  uint32_t u = __builtin_bit_cast(uint32_t, f);
  u += 0x7fffu + ((u >> 16) & 1u);   // round to nearest even
  return (uint16_t)(u >> 16);
}
static __device__ __forceinline__ float bfbits2f(uint16_t b) {
  uint32_t u = ((uint32_t)b) << 16;
  return __builtin_bit_cast(float, u);
}
static __device__ __forceinline__ __bf16 bits2bf(uint16_t b) {
  return __builtin_bit_cast(__bf16, b);
}
static __device__ __forceinline__ uint32_t pack_hilo(float x) {
  uint16_t h = f2bf_bits(x);
  uint16_t l = f2bf_bits(x - bfbits2f(h));
  return ((uint32_t)h << 16) | (uint32_t)l;
}

// ---------------------------------------------------------------------------
// Weight repack kernels: fp32 -> [tap][co][ci] packed hi/lo bf16
// ---------------------------------------------------------------------------
__global__ __launch_bounds__(256) void repack_w1(const float* __restrict__ w1,
                                                 uint32_t* __restrict__ out) {
  int i   = blockIdx.x * 256 + threadIdx.x;   // tap*65536 + co*256 + ci
  int tap = i >> 16;
  int co  = (i >> 8) & 255;
  int ci  = i & 255;
  out[i] = pack_hilo(w1[(co * 256 + ci) * 3 + tap]);
}

__global__ __launch_bounds__(256) void repack_w2(const float* __restrict__ w2,
                                                 uint32_t* __restrict__ out) {
  int i  = blockIdx.x * 256 + threadIdx.x;    // kk*65536 + co*256 + ci
  int kk = i >> 16;
  int co = (i >> 8) & 255;
  int ci = i & 255;
  out[i] = pack_hilo(w2[(co * 256 + ci) * 9 + kk]);
}

// ---------------------------------------------------------------------------
// bf16x3 WMMA step for a 64(M) x 32(N) wave tile.
//   A lds: [M_TILE][K_BLK] rows (hi/lo), B lds: [N_TILE][K_BLK] rows (hi/lo)
//   acc[4][2] v8f
// ---------------------------------------------------------------------------
static __device__ __forceinline__ void mma_step(const __bf16* __restrict__ Ah,
                                                const __bf16* __restrict__ Al,
                                                const __bf16* __restrict__ Bh,
                                                const __bf16* __restrict__ Bl,
                                                int mw, int nw, int lane,
                                                v8f acc[4][2]) {
  const int lm   = lane & 15;
  const int half = lane >> 4;
  // A (16-bit A 16x32): lane<16 -> K[0..7],[16..23]; lane>=16 -> K[8..15],[24..31]
  const int a0 = half ? 8 : 0;
  const int a1 = half ? 24 : 16;
  // B (16-bit B 32x16): lane<16 -> K[0..7],[16..23]; lane>=16 -> K[8..15],[24..31]
  const int b0 = half ? 8 : 0;
  const int b1 = half ? 24 : 16;

  v16bf bh[2], bl[2];
#pragma unroll
  for (int ni = 0; ni < 2; ++ni) {
    const __bf16* ph = Bh + (nw * 32 + ni * 16 + lm) * LDS_STRIDE;
    const __bf16* pl = Bl + (nw * 32 + ni * 16 + lm) * LDS_STRIDE;
    union { v16bf v; v8bf h[2]; } u;
    u.h[0] = *(const v8bf*)(ph + b0); u.h[1] = *(const v8bf*)(ph + b1);
    bh[ni] = u.v;
    u.h[0] = *(const v8bf*)(pl + b0); u.h[1] = *(const v8bf*)(pl + b1);
    bl[ni] = u.v;
  }
#pragma unroll
  for (int mi = 0; mi < 4; ++mi) {
    const __bf16* ph = Ah + (mw * 64 + mi * 16 + lm) * LDS_STRIDE;
    const __bf16* pl = Al + (mw * 64 + mi * 16 + lm) * LDS_STRIDE;
    union { v16bf v; v8bf h[2]; } u;
    v16bf ah, al;
    u.h[0] = *(const v8bf*)(ph + a0); u.h[1] = *(const v8bf*)(ph + a1);
    ah = u.v;
    u.h[0] = *(const v8bf*)(pl + a0); u.h[1] = *(const v8bf*)(pl + a1);
    al = u.v;
#pragma unroll
    for (int ni = 0; ni < 2; ++ni) {
      acc[mi][ni] = __builtin_amdgcn_wmma_f32_16x16x32_bf16(
          false, ah, false, bh[ni], (short)0, acc[mi][ni], false, false);
      acc[mi][ni] = __builtin_amdgcn_wmma_f32_16x16x32_bf16(
          false, al, false, bh[ni], (short)0, acc[mi][ni], false, false);
      acc[mi][ni] = __builtin_amdgcn_wmma_f32_16x16x32_bf16(
          false, ah, false, bl[ni], (short)0, acc[mi][ni], false, false);
    }
  }
}

// ---------------------------------------------------------------------------
// Kernel 1: temporal conv1d (K=3, pad 1) as (up to 3) shifted GEMMs.
// Double-buffered LDS + register prefetch: one barrier per K-step; global
// fetch and LDS staging of step s+1 overlap the 24-WMMA block of step s.
// grid: (25 pixel-blocks, 2 co-blocks, 64 nt); block 256 = 8 waves (2M x 4N)
// ---------------------------------------------------------------------------
__global__ __launch_bounds__(256) void conv1d_wmma(const float* __restrict__ x,
                                                   uint32_t* __restrict__ ws) {
  const uint32_t* __restrict__ w1p = ws + W1P_OFF;
  uint32_t* __restrict__ ypack     = ws + YP_OFF;

  __shared__ __align__(16) __bf16 AhS[2][M_TILE * LDS_STRIDE];
  __shared__ __align__(16) __bf16 AlS[2][M_TILE * LDS_STRIDE];
  __shared__ __align__(16) __bf16 BhS[2][N_TILE * LDS_STRIDE];
  __shared__ __align__(16) __bf16 BlS[2][N_TILE * LDS_STRIDE];

  const int n0    = blockIdx.x * N_TILE;
  const int mBase = blockIdx.y * M_TILE;
  const int nt    = blockIdx.z;
  const int t     = nt & (T_LEN - 1);

  const int tid  = threadIdx.x;
  const int lane = tid & 31;
  const int wv   = tid >> 5;
  const int mw   = wv & 1;   // 2 M-waves of 64
  const int nw   = wv >> 1;  // 4 N-waves of 32

  // valid taps are contiguous: tstart..tstart+ntap-1
  const int tstart = (t == 0) ? 1 : 0;
  const int ntap   = 3 - (t == 0) - (t == T_LEN - 1);
  const int S      = ntap * 8;

  // per-thread staging geometry
  const int arow0 = tid >> 3;           // + j*32
  const int acol0 = (tid & 7) * 4;      // + e
  const int nbase = (tid * 4) & 127;    // B pixel base (+e)
  const int ktb   = tid >> 5;           // B k base (+ j*8)

  uint32_t aReg[16];
  float    bReg[16];

  auto fetchA = [&](int s) {
    const int tap = (s >> 3) + tstart;
    const int kb  = (s & 7) * K_BLK;
    const uint32_t* wt = w1p + tap * 65536 + mBase * 256 + kb;
#pragma unroll
    for (int j = 0; j < 4; ++j) {
      uint4 v = *(const uint4*)(wt + (arow0 + j * 32) * 256 + acol0);
      aReg[j * 4 + 0] = v.x; aReg[j * 4 + 1] = v.y;
      aReg[j * 4 + 2] = v.z; aReg[j * 4 + 3] = v.w;
    }
  };
  auto fetchB = [&](int s) {
    const int tap = (s >> 3) + tstart;
    const int kb  = (s & 7) * K_BLK;
    const int src_nt = nt + tap - 1;
    const float* xp = x + ((size_t)src_nt * C_CH + kb) * HWSZ;
    const int p0 = n0 + nbase;
#pragma unroll
    for (int j = 0; j < 4; ++j) {
      const size_t ko = (size_t)(j * 8 + ktb) * HWSZ;
      if (p0 + 3 < HWSZ) {
        float4 f = *(const float4*)(xp + ko + p0);
        bReg[j * 4 + 0] = f.x; bReg[j * 4 + 1] = f.y;
        bReg[j * 4 + 2] = f.z; bReg[j * 4 + 3] = f.w;
      } else {
#pragma unroll
        for (int e = 0; e < 4; ++e)
          bReg[j * 4 + e] = (p0 + e < HWSZ) ? xp[ko + p0 + e] : 0.f;
      }
    }
  };
  auto stage = [&](int buf) {
    __bf16* __restrict__ Ah = AhS[buf];
    __bf16* __restrict__ Al = AlS[buf];
    __bf16* __restrict__ Bh = BhS[buf];
    __bf16* __restrict__ Bl = BlS[buf];
#pragma unroll
    for (int j = 0; j < 4; ++j) {
      const int row = arow0 + j * 32;
#pragma unroll
      for (int e = 0; e < 4; ++e) {
        uint32_t u = aReg[j * 4 + e];
        Ah[row * LDS_STRIDE + acol0 + e] = bits2bf((uint16_t)(u >> 16));
        Al[row * LDS_STRIDE + acol0 + e] = bits2bf((uint16_t)(u & 0xffffu));
      }
    }
#pragma unroll
    for (int j = 0; j < 4; ++j) {
      const int k = j * 8 + ktb;
#pragma unroll
      for (int e = 0; e < 4; ++e) {
        float f = bReg[j * 4 + e];
        uint16_t h = f2bf_bits(f);
        uint16_t l = f2bf_bits(f - bfbits2f(h));
        Bh[(nbase + e) * LDS_STRIDE + k] = bits2bf(h);
        Bl[(nbase + e) * LDS_STRIDE + k] = bits2bf(l);
      }
    }
  };

  const v8f z = {0.f, 0.f, 0.f, 0.f, 0.f, 0.f, 0.f, 0.f};
  v8f acc[4][2] = {{z, z}, {z, z}, {z, z}, {z, z}};

  fetchA(0); fetchB(0);
  stage(0);
  __syncthreads();
  for (int s = 0; s < S; ++s) {
    const int cur = s & 1;
    if (s + 1 < S) { fetchA(s + 1); fetchB(s + 1); }          // vmem issue
    mma_step(AhS[cur], AlS[cur], BhS[cur], BlS[cur], mw, nw, lane, acc);
    if (s + 1 < S) stage(cur ^ 1);                            // waits on vmem
    __syncthreads();                                          // 1 barrier/step
  }

  // epilogue: write y as packed hi/lo bf16
  const int lm = lane & 15, half = lane >> 4;
#pragma unroll
  for (int mi = 0; mi < 4; ++mi)
#pragma unroll
    for (int ni = 0; ni < 2; ++ni) {
      const int p = n0 + nw * 32 + ni * 16 + lm;
      if (p < HWSZ) {
#pragma unroll
        for (int r = 0; r < 8; ++r) {
          int co = mBase + mw * 64 + mi * 16 + half * 8 + r;
          ypack[((size_t)nt * C_CH + co) * HWSZ + p] = pack_hilo(acc[mi][ni][r]);
        }
      }
    }
}

// ---------------------------------------------------------------------------
// Kernel 2: spatial 3x3 conv (pad 1) as implicit GEMM, K = 9 taps x 256 ci.
// Same double-buffered skeleton; B tile is halo-shifted packed-y with cheap
// precomputed per-thread bounds (only 4 distinct pixels per thread).
// ---------------------------------------------------------------------------
__global__ __launch_bounds__(256) void conv2d_wmma(const uint32_t* __restrict__ ws,
                                                   float* __restrict__ out) {
  const uint32_t* __restrict__ w2p   = ws + W2P_OFF;
  const uint32_t* __restrict__ ypack = ws + YP_OFF;

  __shared__ __align__(16) __bf16 AhS[2][M_TILE * LDS_STRIDE];
  __shared__ __align__(16) __bf16 AlS[2][M_TILE * LDS_STRIDE];
  __shared__ __align__(16) __bf16 BhS[2][N_TILE * LDS_STRIDE];
  __shared__ __align__(16) __bf16 BlS[2][N_TILE * LDS_STRIDE];

  const int n0    = blockIdx.x * N_TILE;
  const int mBase = blockIdx.y * M_TILE;
  const int nt    = blockIdx.z;

  const int tid  = threadIdx.x;
  const int lane = tid & 31;
  const int wv   = tid >> 5;
  const int mw   = wv & 1;
  const int nw   = wv >> 1;

  const int arow0 = tid >> 3;
  const int acol0 = (tid & 7) * 4;
  const int nbase = (tid * 4) & 127;
  const int ktb   = tid >> 5;

  // 4 distinct B pixels per thread: precompute position/bounds once
  int pe[4], he[4], we[4];
#pragma unroll
  for (int e = 0; e < 4; ++e) {
    int p = n0 + nbase + e;
    pe[e] = p;
    he[e] = p / WWID;
    we[e] = p % WWID;
  }

  uint32_t aReg[16];
  uint32_t bReg[16];

  auto fetchA = [&](int s) {
    const int kk = s >> 3;
    const int kb = (s & 7) * K_BLK;
    const uint32_t* wt = w2p + kk * 65536 + mBase * 256 + kb;
#pragma unroll
    for (int j = 0; j < 4; ++j) {
      uint4 v = *(const uint4*)(wt + (arow0 + j * 32) * 256 + acol0);
      aReg[j * 4 + 0] = v.x; aReg[j * 4 + 1] = v.y;
      aReg[j * 4 + 2] = v.z; aReg[j * 4 + 3] = v.w;
    }
  };
  auto fetchB = [&](int s) {
    const int kk = s >> 3;
    const int kb = (s & 7) * K_BLK;
    const int dh = kk / 3 - 1;
    const int dw = kk % 3 - 1;
    const int doff = dh * WWID + dw;
    const uint32_t* yp = ypack + ((size_t)nt * C_CH + kb) * HWSZ;
#pragma unroll
    for (int j = 0; j < 4; ++j) {
      const size_t ko = (size_t)(j * 8 + ktb) * HWSZ;
#pragma unroll
      for (int e = 0; e < 4; ++e) {
        const int hh = he[e] + dh;
        const int ww = we[e] + dw;
        uint32_t u = 0;
        if (hh >= 0 && hh < HH && ww >= 0 && ww < WWID)
          u = yp[ko + (size_t)(pe[e] + doff)];
        bReg[j * 4 + e] = u;
      }
    }
  };
  auto stage = [&](int buf) {
    __bf16* __restrict__ Ah = AhS[buf];
    __bf16* __restrict__ Al = AlS[buf];
    __bf16* __restrict__ Bh = BhS[buf];
    __bf16* __restrict__ Bl = BlS[buf];
#pragma unroll
    for (int j = 0; j < 4; ++j) {
      const int row = arow0 + j * 32;
#pragma unroll
      for (int e = 0; e < 4; ++e) {
        uint32_t u = aReg[j * 4 + e];
        Ah[row * LDS_STRIDE + acol0 + e] = bits2bf((uint16_t)(u >> 16));
        Al[row * LDS_STRIDE + acol0 + e] = bits2bf((uint16_t)(u & 0xffffu));
      }
    }
#pragma unroll
    for (int j = 0; j < 4; ++j) {
      const int k = j * 8 + ktb;
#pragma unroll
      for (int e = 0; e < 4; ++e) {
        uint32_t u = bReg[j * 4 + e];
        Bh[(nbase + e) * LDS_STRIDE + k] = bits2bf((uint16_t)(u >> 16));
        Bl[(nbase + e) * LDS_STRIDE + k] = bits2bf((uint16_t)(u & 0xffffu));
      }
    }
  };

  const v8f z = {0.f, 0.f, 0.f, 0.f, 0.f, 0.f, 0.f, 0.f};
  v8f acc[4][2] = {{z, z}, {z, z}, {z, z}, {z, z}};

  const int S = 9 * 8;
  fetchA(0); fetchB(0);
  stage(0);
  __syncthreads();
  for (int s = 0; s < S; ++s) {
    const int cur = s & 1;
    if (s + 1 < S) { fetchA(s + 1); fetchB(s + 1); }
    mma_step(AhS[cur], AlS[cur], BhS[cur], BlS[cur], mw, nw, lane, acc);
    if (s + 1 < S) stage(cur ^ 1);
    __syncthreads();
  }

  const int lm = lane & 15, half = lane >> 4;
#pragma unroll
  for (int mi = 0; mi < 4; ++mi)
#pragma unroll
    for (int ni = 0; ni < 2; ++ni) {
      const int p = n0 + nw * 32 + ni * 16 + lm;
      if (p < HWSZ) {
#pragma unroll
        for (int r = 0; r < 8; ++r) {
          int co = mBase + mw * 64 + mi * 16 + half * 8 + r;
          out[((size_t)nt * C_CH + co) * HWSZ + p] = acc[mi][ni][r];
        }
      }
    }
}

// ---------------------------------------------------------------------------
extern "C" void kernel_launch(void* const* d_in, const int* in_sizes, int n_in,
                              void* d_out, int out_size, void* d_ws, size_t ws_size,
                              hipStream_t stream) {
  (void)in_sizes; (void)n_in; (void)out_size; (void)ws_size;
  const float* x  = (const float*)d_in[0];   // (64, 256, 56, 56)
  const float* w1 = (const float*)d_in[1];   // (256, 256, 3)
  const float* w2 = (const float*)d_in[2];   // (256, 256, 3, 3)
  float*    out   = (float*)d_out;           // (64, 256, 56, 56)
  uint32_t* ws    = (uint32_t*)d_ws;

  repack_w1<<<3 * 256 * 256 / 256, 256, 0, stream>>>(w1, ws + W1P_OFF);
  repack_w2<<<9 * 256 * 256 / 256, 256, 0, stream>>>(w2, ws + W2P_OFF);

  conv1d_wmma<<<dim3(N_BLOCKS, C_CH / M_TILE, NT_TOT), 256, 0, stream>>>(x, ws);
  conv2d_wmma<<<dim3(N_BLOCKS, C_CH / M_TILE, NT_TOT), 256, 0, stream>>>(ws, out);
}
